// ApproxRepSet_59828894433555
// MI455X (gfx1250) — matrix operand order repaired and checked by reading, not compiled, not run on hardware
//
#include <hip/hip_runtime.h>

typedef __attribute__((ext_vector_type(16))) _Float16 v16h;
typedef __attribute__((ext_vector_type(8)))  _Float16 v8h;
typedef __attribute__((ext_vector_type(8)))  float    v8f;
typedef __attribute__((ext_vector_type(4)))  unsigned int u32x4;
typedef __attribute__((ext_vector_type(8)))  int      i32x8;
typedef __attribute__((ext_vector_type(4)))  int      i32x4;

#define WC_STRIDE   72                         // halves per column: 64 data + 8 pad
#define WC_BYTES    (512 * WC_STRIDE * 2)      // 73728 bytes, contiguous f16 image
#define WC_QWORDS   (WC_BYTES / 8)             // 9216 8-byte elements

// ---------------- kernel 0: zero accumulator ----------------
__global__ void repset_zero(float* __restrict__ acc, int n) {
  int i = blockIdx.x * blockDim.x + threadIdx.x;
  if (i < n) acc[i] = 0.0f;
}

// ---------------- kernel 1: one-time Wc f32 -> f16 column-major padded image ----
__global__ void repset_cvt_wc(const float* __restrict__ Wc, _Float16* __restrict__ wc16) {
  int i = blockIdx.x * blockDim.x + threadIdx.x;   // 0 .. 32767
  if (i < 64 * 512) {
    int d = i >> 9;        // K row 0..63
    int k = i & 511;       // N col 0..511
    wc16[k * WC_STRIDE + d] = (_Float16)Wc[i];
  }
}

// A fragment loader: lane holds 16 consecutive K values (k = 16h + s), matching B.
__device__ inline v16h load_a_frag(const float* __restrict__ xrow, int c1) {
  float4 f0 = *(const float4*)(xrow + c1 + 0);
  float4 f1 = *(const float4*)(xrow + c1 + 4);
  float4 f2 = *(const float4*)(xrow + c1 + 8);
  float4 f3 = *(const float4*)(xrow + c1 + 12);
  v16h a;
  a[0]  = (_Float16)f0.x; a[1]  = (_Float16)f0.y; a[2]  = (_Float16)f0.z; a[3]  = (_Float16)f0.w;
  a[4]  = (_Float16)f1.x; a[5]  = (_Float16)f1.y; a[6]  = (_Float16)f1.z; a[7]  = (_Float16)f1.w;
  a[8]  = (_Float16)f2.x; a[9]  = (_Float16)f2.y; a[10] = (_Float16)f2.z; a[11] = (_Float16)f2.w;
  a[12] = (_Float16)f3.x; a[13] = (_Float16)f3.y; a[14] = (_Float16)f3.z; a[15] = (_Float16)f3.w;
  return a;
}

// B fragment: lane holds 16 consecutive K values of one column.
__device__ inline v16h load_b_frag(const _Float16* __restrict__ p) {
  v8h lo = *(const v8h*)(p);
  v8h hi = *(const v8h*)(p + 8);
  return __builtin_shufflevector(lo, hi, 0,1,2,3,4,5,6,7,8,9,10,11,12,13,14,15);
}

// ---------------- kernel 2: GEMM (f16 WMMA) + relu + max-over-E + sum-over-P ----
__global__ __launch_bounds__(256)
void repset_gemm_reduce(const float* __restrict__ X,        // [B*P, 64]
                        const _Float16* __restrict__ wc16,  // padded f16 image, WC_BYTES
                        float* __restrict__ accum) {        // [B, 32]
  extern __shared__ _Float16 wcLds[];                       // WC_BYTES of LDS
  const int tid = threadIdx.x;

  // --- Stage Wc image into LDS with one Tensor Data Mover DMA (wave 0 only) ---
  if (tid < 32) {
    unsigned long long ga = (unsigned long long)wc16;
    unsigned ldsOff = (unsigned)__builtin_amdgcn_groupstaticsize();  // dynamic LDS base
    u32x4 g0;
    g0[0] = 1u;                                        // count=1, user descriptor
    g0[1] = ldsOff;                                    // lds_addr (bytes)
    g0[2] = (unsigned)(ga & 0xFFFFFFFFu);              // global_addr[31:0]
    g0[3] = (unsigned)((ga >> 32) & 0x01FFFFFFu)       // global_addr[56:32]
          | (2u << 30);                                // type = 2 ("image")
    i32x8 g1;
    g1[0] = 3 << 16;                                   // data_size = 3 (8 bytes)
    g1[1] = (int)((WC_QWORDS & 0xFFFF) << 16);         // tensor_dim0[15:0]
    g1[2] = (int)(((WC_QWORDS >> 16) & 0xFFFF)         // tensor_dim0[31:16]
          | (1 << 16));                                // tensor_dim1 = 1
    g1[3] = (int)((WC_QWORDS & 0xFFFF) << 16);         // tile_dim0 = 9216
    g1[4] = 1;                                         // tile_dim1 = 1, tile_dim2 = 0
    g1[5] = WC_QWORDS;                                 // tensor_dim0_stride lo32
    g1[6] = 0;                                         // stride hi / dim1_stride lo
    g1[7] = 0;
    i32x4 z4 = {};
    i32x8 z8 = {};
    __builtin_amdgcn_tensor_load_to_lds(g0, g1, z4, z4, z8, 0);
    __builtin_amdgcn_s_wait_tensorcnt(0);
  }
  __syncthreads();

  const int wave = tid >> 5;
  const int lane = tid & 31;
  const int m    = lane & 15;
  const int hi   = lane >> 4;
  const int rowBase = blockIdx.x * 128 + wave * 16;     // flattened (b,p) row
  const int b = rowBase >> 10;                          // P = 1024

  const float* xrow = X + (size_t)(rowBase + m) * 64;

  // A fragments for K = [0,32) and [32,64): reused across all 32 N-tiles.
  const v16h a0 = load_a_frag(xrow,  0 + hi * 16);
  const v16h a1 = load_a_frag(xrow, 32 + hi * 16);

  // Running max accumulators (C/D layout). Init 0 => relu folded into fmaxf.
  v8f hmax0 = {};
  v8f hmax1 = {};

  #pragma unroll 4
  for (int e = 0; e < 16; ++e) {
    // h-tile 0 -> columns e*32 + [0,16), h-tile 1 -> columns e*32 + [16,32)
    const _Float16* base0 = wcLds + (e * 32 + m) * WC_STRIDE + hi * 16;
    const _Float16* base1 = base0 + 16 * WC_STRIDE;
    const v16h b00 = load_b_frag(base0);        // ht0, K  0..31
    const v16h b01 = load_b_frag(base0 + 32);   // ht0, K 32..63
    const v16h b10 = load_b_frag(base1);        // ht1, K  0..31
    const v16h b11 = load_b_frag(base1 + 32);   // ht1, K 32..63
    v8f acc0 = {};
    v8f acc1 = {};
    acc0 = __builtin_amdgcn_wmma_f32_16x16x32_f16(false, a0, false, b00,
                                                  (short)0, acc0, false, false);
    acc1 = __builtin_amdgcn_wmma_f32_16x16x32_f16(false, a0, false, b10,
                                                  (short)0, acc1, false, false);
    acc0 = __builtin_amdgcn_wmma_f32_16x16x32_f16(false, a1, false, b01,
                                                  (short)0, acc0, false, false);
    acc1 = __builtin_amdgcn_wmma_f32_16x16x32_f16(false, a1, false, b11,
                                                  (short)0, acc1, false, false);
    #pragma unroll
    for (int j = 0; j < 8; ++j) hmax0[j] = fmaxf(hmax0[j], acc0[j]);
    #pragma unroll
    for (int j = 0; j < 8; ++j) hmax1[j] = fmaxf(hmax1[j], acc1[j]);
  }

  // Sum the 16 rows of this wave's M-tile (all same batch b).
  float s0 = (hmax0[0] + hmax0[1]) + (hmax0[2] + hmax0[3])
           + (hmax0[4] + hmax0[5]) + (hmax0[6] + hmax0[7]);
  float s1 = (hmax1[0] + hmax1[1]) + (hmax1[2] + hmax1[3])
           + (hmax1[4] + hmax1[5]) + (hmax1[6] + hmax1[7]);
  s0 += __shfl_xor(s0, 16, 32);
  s1 += __shfl_xor(s1, 16, 32);

  if (hi == 0) {
    atomicAdd(&accum[b * 32 +      m], s0);   // h = m
    atomicAdd(&accum[b * 32 + 16 + m], s1);   // h = 16 + m
  }
}

// ---------------- kernel 3: tiny MLP tail, one thread per batch row ----------
__global__ __launch_bounds__(128)
void repset_mlp(const float* __restrict__ accum,
                const float* __restrict__ w1, const float* __restrict__ b1,
                const float* __restrict__ w2, const float* __restrict__ b2,
                const float* __restrict__ w3, const float* __restrict__ b3,
                float* __restrict__ out) {
  const int bi = threadIdx.x;
  float s[32];
  #pragma unroll
  for (int i = 0; i < 32; ++i) s[i] = accum[bi * 32 + i];

  float h1[64];
  for (int j = 0; j < 64; ++j) {
    float v = b1[j];
    #pragma unroll
    for (int i = 0; i < 32; ++i) v += s[i] * w1[i * 64 + j];
    h1[j] = fmaxf(v, 0.0f);
  }
  float h2[64];
  for (int j = 0; j < 64; ++j) {
    float v = b2[j];
    #pragma unroll
    for (int i = 0; i < 64; ++i) v += h1[i] * w2[i * 64 + j];
    h2[j] = fmaxf(v, 0.0f);
  }
  for (int j = 0; j < 10; ++j) {
    float v = b3[j];
    #pragma unroll
    for (int i = 0; i < 64; ++i) v += h2[i] * w3[i * 10 + j];
    out[bi * 10 + j] = v;
  }
}

extern "C" void kernel_launch(void* const* d_in, const int* in_sizes, int n_in,
                              void* d_out, int out_size, void* d_ws, size_t ws_size,
                              hipStream_t stream) {
  (void)in_sizes; (void)n_in; (void)out_size; (void)ws_size;
  const float* X  = (const float*)d_in[0];   // [128,1024,64]
  const float* Wc = (const float*)d_in[1];   // [64,512]
  const float* w1 = (const float*)d_in[2];   // [32,64]
  const float* b1 = (const float*)d_in[3];   // [64]
  const float* w2 = (const float*)d_in[4];   // [64,64]
  const float* b2 = (const float*)d_in[5];   // [64]
  const float* w3 = (const float*)d_in[6];   // [64,10]
  const float* b3 = (const float*)d_in[7];   // [10]
  float* out   = (float*)d_out;              // [128,10]

  // ws layout: [0, 16KB) accum f32[128*32]; [16KB, 16KB+WC_BYTES) f16 Wc image
  float*    accum = (float*)d_ws;
  _Float16* wc16  = (_Float16*)((char*)d_ws + 16384);

  repset_zero<<<16, 256, 0, stream>>>(accum, 128 * 32);
  repset_cvt_wc<<<128, 256, 0, stream>>>(Wc, wc16);

  // 131072 rows / (8 waves * 16 rows) = 1024 workgroups; 72 KB dynamic LDS
  repset_gemm_reduce<<<1024, 256, WC_BYTES, stream>>>(X, wc16, accum);

  repset_mlp<<<1, 128, 0, stream>>>(accum, w1, b1, w2, b2, w3, b3, out);
}
